// OWXRA_51488067945159
// MI455X (gfx1250) — matrix operand order repaired and checked by reading, not compiled, not run on hardware
//
#include <hip/hip_runtime.h>
#include <hip/hip_bf16.h>
#include <cstdint>

// ---------------------------------------------------------------------------
// Overlapping-window attention (Swin-upscale style), CDNA5 / gfx1250, wave32.
// Shapes: B=2, H=W=128, C=96, WS=16, OWIN=24, PAD=4, WSU=32
//   kv   = x @ Wkv + bkv            -> padded (2,136,136,192) f16   [WMMA]
//   attn = softmax(q k^T * s + bias)-> flash, per window            [WMMA]
//   out  = (attn v) @ Wproj + bproj -> (128,1024,96) f32            [WMMA]
// rpi bias indices computed arithmetically: idx = 55*(ki-qi+9)+(kj-qj+9) mod 3025
// K-chunk staging uses GLOBAL_LOAD_ASYNC_TO_LDS_B128 (ASYNCcnt) on gfx1250.
// ---------------------------------------------------------------------------

typedef __attribute__((ext_vector_type(16))) _Float16 v16h;
typedef __attribute__((ext_vector_type(8)))  float    v8f;

union V16H { v16h v; _Float16 h[16]; uint4 q[2]; };

#define WMMA_F16(a, b, c) \
  __builtin_amdgcn_wmma_f32_16x16x32_f16(false, (a), false, (b), (short)0, (c), false, false)

// ---------------------------------------------------------------------------
// Kernel 0: zero the padded KV buffer (pad border must contribute zeros).
// ---------------------------------------------------------------------------
__global__ void zero_kernel(uint4* __restrict__ p, int n) {
  const int i = blockIdx.x * blockDim.x + threadIdx.x;
  if (i < n) p[i] = uint4{0u, 0u, 0u, 0u};
}

// ---------------------------------------------------------------------------
// Kernel 1: kv = x @ Wkv + bkv, written f16 into padded (2,136,136,192).
// 8 waves/block, each wave: 16 pixels x 192 outputs, K=96 (3 WMMA chunks).
// ---------------------------------------------------------------------------
__global__ __launch_bounds__(256) void kv_kernel(
    const float* __restrict__ x, const float* __restrict__ Wkv,
    const float* __restrict__ bkv, _Float16* __restrict__ kvp) {
  __shared__ __align__(16) _Float16 wT[192 * 96];  // Wkv transposed: [n][k]
  const int tid = threadIdx.x;
  for (int i = tid; i < 192 * 96; i += 256) {
    const int n = i / 96, k = i - n * 96;
    wT[i] = (_Float16)Wkv[k * 192 + n];
  }
  __syncthreads();

  const int wv = tid >> 5, l = tid & 31;
  const int lane16 = l & 15, hi = l >> 4;
  const int tb = blockIdx.x * 128 + wv * 16;  // first pixel of this wave's tile

  // A tile: 16 pixels x 96 (f32 -> f16), contiguous K layout (k = 16*hi + e)
  const float* xr = x + (size_t)(tb + lane16) * 96;
  V16H a[3];
#pragma unroll
  for (int c = 0; c < 3; ++c) {
    const int kb = c * 32 + hi * 16;
#pragma unroll
    for (int e = 0; e < 8; ++e) a[c].h[e] = (_Float16)xr[kb + e];
#pragma unroll
    for (int e = 0; e < 8; ++e) a[c].h[8 + e] = (_Float16)xr[kb + 8 + e];
  }

#pragma unroll
  for (int t = 0; t < 12; ++t) {
    const int n = t * 16 + lane16;
    const float bv = bkv[n];
    v8f cc;
#pragma unroll
    for (int r = 0; r < 8; ++r) cc[r] = bv;
#pragma unroll
    for (int c = 0; c < 3; ++c) {
      const v16h b = *(const v16h*)(&wT[n * 96 + c * 32 + hi * 16]);
      cc = WMMA_F16(a[c].v, b, cc);
    }
#pragma unroll
    for (int r = 0; r < 8; ++r) {
      const int p2 = tb + r + hi * 8;           // pixel of C-row r
      const int b2 = p2 >> 14, rem = p2 & 16383;
      const int i2 = rem >> 7, j2 = rem & 127;
      kvp[(((size_t)(b2 * 136 + i2 + 4)) * 136 + (j2 + 4)) * 192 + n] =
          (_Float16)cc[r];
    }
  }
}

// ---------------------------------------------------------------------------
// Kernel 2: flash attention per (window, Q-tile). 1024 blocks, 256 threads.
//   K half of each 32-key chunk staged with async global->LDS copies
//   (ASYNCcnt); V half loaded + transposed into [d][key] so the P*V B-tiles
//   are contiguous-16 reads. Bias table in LDS; rpi computed arithmetically.
//   Online softmax; P restaged via LDS to A-layout for the P*V WMMA.
// ---------------------------------------------------------------------------
__global__ __launch_bounds__(256) void attn_kernel(
    const float* __restrict__ x2, const float* __restrict__ bias_table,
    const _Float16* __restrict__ kvp, _Float16* __restrict__ attnh) {
  __shared__ float blds[3040];                       // 3025 used
  __shared__ __align__(16) _Float16 klds[32 * 96];   // [key][d]
  __shared__ __align__(16) _Float16 vT[96 * 32];     // [d][key]
  __shared__ __align__(16) _Float16 plds[8 * 16 * 32];  // per-wave P stage

  const int tid = threadIdx.x;
  for (int i = tid; i < 3025; i += 256) blds[i] = bias_table[i];

  const int blk = blockIdx.x;
  const int w = blk >> 3, qt = blk & 7;
  const int bb = w >> 6, wh = (w >> 3) & 7, ww = w & 7;
  const int wv = tid >> 5, l = tid & 31;
  const int lane16 = l & 15, hi = l >> 4;
  const int qbase = qt * 128 + wv * 16;

  // Q tile (scaled), A layout, kept in registers for all 18 K-chunks.
  const float scale = 0.1020620726159658f;  // 96^-0.5
  const float* qrow = x2 + ((size_t)w * 1024 + qbase + lane16) * 96;
  V16H qa[3];
#pragma unroll
  for (int c = 0; c < 3; ++c) {
    const int kb = c * 32 + hi * 16;
#pragma unroll
    for (int e = 0; e < 8; ++e) qa[c].h[e] = (_Float16)(qrow[kb + e] * scale);
#pragma unroll
    for (int e = 0; e < 8; ++e)
      qa[c].h[8 + e] = (_Float16)(qrow[kb + 8 + e] * scale);
  }

  float mrow[8], lrow[8];
#pragma unroll
  for (int r = 0; r < 8; ++r) { mrow[r] = -1e30f; lrow[r] = 0.0f; }
  v8f acc[6];
#pragma unroll
  for (int t = 0; t < 6; ++t) acc[t] = v8f{};

  const _Float16* winBase =
      kvp + (((size_t)(bb * 136 + wh * 16)) * 136 + ww * 16) * 192;
  _Float16* pw = &plds[wv * 512];
  // LDS byte offset of klds (flat shared pointer low 32 bits = LDS address).
  const uint32_t kldsBase = (uint32_t)(uintptr_t)&klds[0];

  for (int kc = 0; kc < 18; ++kc) {
    __syncthreads();  // previous chunk fully consumed

    // K half: CDNA5 async global->LDS b128 copies, no VGPR staging.
    for (int it = tid; it < 384; it += 256) {
      const int hb = it * 8;                    // half-index into [key][96]
      const int key = hb / 96, ch = hb - key * 96;
      const int gk = kc * 32 + key;
      const int ki = gk / 24, kj = gk - ki * 24;
      const _Float16* g = winBase + ((size_t)(ki * 136) + kj) * 192 + ch;
      const uint32_t ldsa = kldsBase + (uint32_t)(hb * 2);
      asm volatile("global_load_async_to_lds_b128 %0, %1, off"
                   :: "v"(ldsa), "v"(g) : "memory");
    }
    // V half: load + transposed scatter into vT ([d][key]).
    for (int it = tid; it < 384; it += 256) {
      const int hb = it * 8;
      const int key = hb / 96, ch = hb - key * 96;
      const int gk = kc * 32 + key;
      const int ki = gk / 24, kj = gk - ki * 24;
      const uint4 d =
          *(const uint4*)(winBase + ((size_t)(ki * 136) + kj) * 192 + 96 + ch);
      const _Float16* hp = (const _Float16*)&d;
#pragma unroll
      for (int j = 0; j < 8; ++j) vT[(ch + j) * 32 + key] = hp[j];
    }
    if (kc + 1 < 18) {  // gfx1250 global_prefetch_b8 for next chunk
      const int gkn = (kc + 1) * 32;
      __builtin_prefetch(
          winBase + ((size_t)((gkn / 24) * 136) + (gkn % 24)) * 192, 0, 1);
    }
    asm volatile("s_wait_asynccnt 0x0" ::: "memory");
    __syncthreads();

    // S = q k^T for two 16-key subtiles (K-dim = d = 96, 3 WMMA each).
    v8f s0 = v8f{}, s1 = v8f{};
#pragma unroll
    for (int c = 0; c < 3; ++c) {
      const int ks = c * 32 + hi * 16;
      const v16h b0 = *(const v16h*)(&klds[lane16 * 96 + ks]);
      const v16h b1 = *(const v16h*)(&klds[(16 + lane16) * 96 + ks]);
      s0 = WMMA_F16(qa[c].v, b0, s0);
      s1 = WMMA_F16(qa[c].v, b1, s1);
    }

    // Bias (arithmetic rpi) + online softmax. Row r lives in lanes of one
    // 16-lane half (C layout), so shfl_xor 1/2/4/8 reduces a row.
    const int key0 = kc * 32 + lane16, key1 = key0 + 16;
    const int ki0 = key0 / 24, kj0 = key0 - ki0 * 24;
    const int ki1 = key1 / 24, kj1 = key1 - ki1 * 24;
#pragma unroll
    for (int r = 0; r < 8; ++r) {
      const int m = r + hi * 8;
      const int qq = qbase + m;
      const int qi = qq >> 5, qj = qq & 31;
      int v0 = 55 * (ki0 - qi + 9) + (kj0 - qj + 9);
      if (v0 < 0) v0 += 3025;
      int v1 = 55 * (ki1 - qi + 9) + (kj1 - qj + 9);
      if (v1 < 0) v1 += 3025;
      const float t0 = s0[r] + blds[v0];
      const float t1 = s1[r] + blds[v1];
      float cmax = fmaxf(t0, t1);
      cmax = fmaxf(cmax, __shfl_xor(cmax, 1, 32));
      cmax = fmaxf(cmax, __shfl_xor(cmax, 2, 32));
      cmax = fmaxf(cmax, __shfl_xor(cmax, 4, 32));
      cmax = fmaxf(cmax, __shfl_xor(cmax, 8, 32));
      const float mnew = fmaxf(mrow[r], cmax);
      const float f = __expf(mrow[r] - mnew);
      mrow[r] = mnew;
      const float p0 = __expf(t0 - mnew);
      const float p1 = __expf(t1 - mnew);
      float ps = p0 + p1;
      ps += __shfl_xor(ps, 1, 32);
      ps += __shfl_xor(ps, 2, 32);
      ps += __shfl_xor(ps, 4, 32);
      ps += __shfl_xor(ps, 8, 32);
      lrow[r] = lrow[r] * f + ps;
#pragma unroll
      for (int t = 0; t < 6; ++t) acc[t][r] *= f;
      pw[m * 32 + lane16] = (_Float16)p0;        // stage P (C->A relayout)
      pw[m * 32 + 16 + lane16] = (_Float16)p1;
    }
    __builtin_amdgcn_wave_barrier();  // LDS per-wave in-order; block reorder

    // O += P * V  (A = P 16x32, B = V^T tiles 32x16, 6 output N-tiles).
    V16H pa;
    pa.q[0] = *(const uint4*)(pw + lane16 * 32 + hi * 16);
    pa.q[1] = *(const uint4*)(pw + lane16 * 32 + hi * 16 + 8);
#pragma unroll
    for (int t = 0; t < 6; ++t) {
      const v16h b = *(const v16h*)(&vT[(t * 16 + lane16) * 32 + hi * 16]);
      acc[t] = WMMA_F16(pa.v, b, acc[t]);
    }
  }

  // Normalize and emit f16 intermediate for the projection GEMM.
#pragma unroll
  for (int r = 0; r < 8; ++r) lrow[r] = 1.0f / lrow[r];
  _Float16* orow = attnh + ((size_t)w * 1024 + qbase) * 96;
#pragma unroll
  for (int t = 0; t < 6; ++t) {
    const int n = t * 16 + lane16;
#pragma unroll
    for (int r = 0; r < 8; ++r) {
      const int m = r + hi * 8;
      orow[(size_t)m * 96 + n] = (_Float16)(acc[t][r] * lrow[r]);
    }
  }
}

// ---------------------------------------------------------------------------
// Kernel 3: out = attnh @ Wproj + bproj  (131072 x 96 x 96), f32 output.
// ---------------------------------------------------------------------------
__global__ __launch_bounds__(256) void proj_kernel(
    const _Float16* __restrict__ attnh, const float* __restrict__ Wproj,
    const float* __restrict__ bproj, float* __restrict__ out) {
  __shared__ __align__(16) _Float16 wT[96 * 96];  // Wproj transposed [n][k]
  const int tid = threadIdx.x;
  for (int i = tid; i < 96 * 96; i += 256) {
    const int n = i / 96, k = i - n * 96;
    wT[i] = (_Float16)Wproj[k * 96 + n];
  }
  __syncthreads();

  const int wv = tid >> 5, l = tid & 31;
  const int lane16 = l & 15, hi = l >> 4;
  const int tb = blockIdx.x * 128 + wv * 16;

  const _Float16* ar = attnh + (size_t)(tb + lane16) * 96;
  V16H a[3];
#pragma unroll
  for (int c = 0; c < 3; ++c) {
    a[c].q[0] = *(const uint4*)(ar + c * 32 + hi * 16);
    a[c].q[1] = *(const uint4*)(ar + c * 32 + hi * 16 + 8);
  }
#pragma unroll
  for (int t = 0; t < 6; ++t) {
    const int n = t * 16 + lane16;
    const float bv = bproj[n];
    v8f cc;
#pragma unroll
    for (int r = 0; r < 8; ++r) cc[r] = bv;
#pragma unroll
    for (int c = 0; c < 3; ++c) {
      const v16h b = *(const v16h*)(&wT[n * 96 + c * 32 + hi * 16]);
      cc = WMMA_F16(a[c].v, b, cc);
    }
#pragma unroll
    for (int r = 0; r < 8; ++r)
      out[(size_t)(tb + r + hi * 8) * 96 + n] = cc[r];
  }
}

// ---------------------------------------------------------------------------
extern "C" void kernel_launch(void* const* d_in, const int* in_sizes, int n_in,
                              void* d_out, int out_size, void* d_ws,
                              size_t ws_size, hipStream_t stream) {
  (void)in_sizes; (void)n_in; (void)out_size; (void)ws_size;
  const float* x          = (const float*)d_in[0];  // (2,128,128,96)
  const float* x2         = (const float*)d_in[1];  // (128,1024,96)
  const float* Wkv        = (const float*)d_in[2];  // (96,192)
  const float* bkv        = (const float*)d_in[3];  // (192,)
  const float* bias_table = (const float*)d_in[4];  // (3025,1)
  const float* Wproj      = (const float*)d_in[5];  // (96,96)
  const float* bproj      = (const float*)d_in[6];  // (96,)
  float* out = (float*)d_out;                       // (128,1024,96)

  // Workspace: padded KV (f16) then attention intermediate (f16).
  _Float16* kvp = (_Float16*)d_ws;
  const size_t kvpBytes = (size_t)2 * 136 * 136 * 192 * sizeof(_Float16);
  _Float16* attnh = (_Float16*)((char*)d_ws + kvpBytes);

  zero_kernel<<<3468, 256, 0, stream>>>((uint4*)kvp, 887808);
  kv_kernel<<<256, 256, 0, stream>>>(x, Wkv, bkv, kvp);
  attn_kernel<<<1024, 256, 0, stream>>>(x2, bias_table, kvp, attnh);
  proj_kernel<<<1024, 256, 0, stream>>>(attnh, Wproj, bproj, out);
}
